// PlanNet_10857677324419
// MI455X (gfx1250) — matrix operand order repaired and checked by reading, not compiled, not run on hardware
//
#include <hip/hip_runtime.h>
#include <stdint.h>
#include <math.h>

typedef __attribute__((ext_vector_type(16))) __bf16 v16bf;
typedef __attribute__((ext_vector_type(2)))  __bf16 v2bf;
typedef __attribute__((ext_vector_type(8)))  float  v8f;

#define N_PATHS_C 20000
#define L_C 16
#define N_LINKS_C 10000
#define N_NODES_C 2000

// ---------------- helpers ----------------

static __device__ __forceinline__ unsigned short f2bf(float f) {
  union { float f; unsigned u; } x; x.f = f;
  unsigned u = x.u;
  u += 0x7fffu + ((u >> 16) & 1u);           // round-to-nearest-even
  return (unsigned short)(u >> 16);
}

// pack two f32 into packed bf16 pair (hardware v_cvt_pk_bf16_f32 when available)
static __device__ __forceinline__ unsigned pk2bf(float lo, float hi) {
#if __has_builtin(__builtin_amdgcn_cvt_pk_bf16_f32)
  union { v2bf v; unsigned u; } x;
  x.v = __builtin_amdgcn_cvt_pk_bf16_f32(lo, hi);
  return x.u;
#else
  return (unsigned)f2bf(lo) | ((unsigned)f2bf(hi) << 16);
#endif
}

union FragB16 { v16bf v; unsigned u[8]; };

// A fragment (16x32 bf16) from LDS row of bf16 (row = M = lane%16).
// elems i<8 -> K = kbase+i ; i>=8 -> K = kbase+16+(i-8), kbase = kt*32 + 8*(lane/16)
static __device__ __forceinline__ v16bf load_a_lds(const unsigned short* row, int kbase) {
  FragB16 a;
#pragma unroll
  for (int j = 0; j < 4; ++j) a.u[j]     = *reinterpret_cast<const unsigned*>(row + kbase + 2 * j);
#pragma unroll
  for (int j = 0; j < 4; ++j) a.u[4 + j] = *reinterpret_cast<const unsigned*>(row + kbase + 16 + 2 * j);
  return a.v;
}

// packed fragment from global: layout [tile][lane][16] bf16, 32B contiguous per lane
static __device__ __forceinline__ v16bf load_frag_glb(const unsigned short* __restrict__ P,
                                                      long tileIdx, int lane) {
  FragB16 b;
  const unsigned* p = reinterpret_cast<const unsigned*>(P + (tileIdx * 32 + lane) * 16);
#pragma unroll
  for (int j = 0; j < 8; ++j) b.u[j] = p[j];
  return b.v;
}

static __device__ __forceinline__ v8f wmma_bf16(v16bf a, v16bf b, v8f c) {
  return __builtin_amdgcn_wmma_f32_16x16x32_bf16(false, a, false, b, (short)0, c, false, false);
}

static __device__ __forceinline__ v8f vzero8() {
  v8f z = {0.f, 0.f, 0.f, 0.f, 0.f, 0.f, 0.f, 0.f};
  return z;
}

static __device__ __forceinline__ v8f gemm_acc(const unsigned short* sArow, int kTiles,
                                               const unsigned short* __restrict__ Wp,
                                               int Nt, int nt, int lane, int hlf, v8f acc) {
#pragma unroll 2
  for (int kt = 0; kt < kTiles; ++kt) {
    v16bf a = load_a_lds(sArow, kt * 32 + 8 * hlf);
    v16bf b = load_frag_glb(Wp, (long)kt * Nt + nt, lane);
    acc = wmma_bf16(a, b, acc);
  }
  return acc;
}

// ---- CDNA5 async global->LDS copy (ASYNCcnt path, per-lane addresses) ----
static __device__ __forceinline__ void async_copy16(const float* g, const float* l) {
  unsigned lds = (unsigned)(size_t)(__attribute__((address_space(3))) const void*)l;
  asm volatile("global_load_async_to_lds_b128 %0, %1, off" :: "v"(lds), "v"(g) : "memory");
}
static __device__ __forceinline__ void wait_async0() {
  asm volatile("s_wait_asynccnt 0x0" ::: "memory");
}

// ---------------- utility kernels ----------------

__global__ void k_zero_f32(float* p, int n) {
  int i = blockIdx.x * blockDim.x + threadIdx.x;
  if (i < n) p[i] = 0.f;
}

__global__ void k_init_vec_state(const float* __restrict__ init, float* __restrict__ st, int rows) {
  int i = blockIdx.x * blockDim.x + threadIdx.x;
  if (i < rows * 128) st[i] = ((i & 127) == 0) ? init[i >> 7] : 0.f;
}

__global__ void k_init_path_state(const float* __restrict__ pinit, float* __restrict__ st) {
  int i = blockIdx.x * blockDim.x + threadIdx.x;
  if (i < N_PATHS_C * 128) {
    int r = i >> 7, c = i & 127;
    st[i] = (c == 0) ? pinit[r] : (c == 1) ? pinit[N_PATHS_C + r] : 0.f;
  }
}

// Pack row-major f32 W[K][N] into B-fragment layout [kt][nt][lane][16] bf16
__global__ void k_pack_b(const float* __restrict__ W, unsigned short* __restrict__ out,
                         int Kt, int Nt, int Kreal, int N) {
  int idx = blockIdx.x * blockDim.x + threadIdx.x;
  if (idx >= Kt * Nt * 32) return;
  int lane = idx & 31, tile = idx >> 5;
  int nt = tile % Nt, kt = tile / Nt;
  int n = nt * 16 + (lane & 15);
  int kb = kt * 32 + 8 * (lane >> 4);
  unsigned short* o = out + (size_t)idx * 16;
#pragma unroll
  for (int i = 0; i < 16; ++i) {
    int k = kb + ((i < 8) ? i : (i + 8));
    float v = (k < Kreal) ? W[(size_t)k * N + n] : 0.f;
    o[i] = f2bf(v);
  }
}

// Pack laplacian [2000][2000] into A-fragment layout [mt][kt][lane][16] bf16 (K padded to 2016)
__global__ void k_pack_a_lap(const float* __restrict__ lap, unsigned short* __restrict__ out) {
  int idx = blockIdx.x * blockDim.x + threadIdx.x;
  if (idx >= 125 * 63 * 32) return;
  int lane = idx & 31, tile = idx >> 5;
  int kt = tile % 63, mt = tile / 63;
  int M = mt * 16 + (lane & 15);
  int hlf = lane >> 4;
  unsigned short* o = out + (size_t)idx * 16;
#pragma unroll
  for (int i = 0; i < 16; ++i) {
    int k = kt * 32 + 8 * hlf + ((i < 8) ? i : (i + 8));
    float v = (k < N_NODES_C) ? lap[(size_t)M * N_NODES_C + k] : 0.f;
    o[i] = f2bf(v);
  }
}

// ------- fused GRU scan: async double-buffered gather + 16-step GRU + m scatter -------
// 384 threads = 12 waves; each wave owns 2 of the 24 output n-tiles, with its
// weight slice (2*(8+4) fragments = 192 VGPRs) preloaded in registers.
// x lands f32 via async DMA, is converted to bf16 ONCE per step (pk-convert),
// then all waves build A-fragments with pure ds_load_b32 (no VALU on the path).

__global__ __launch_bounds__(384) void k_gru_scan(
    const float* __restrict__ link_state, const float* __restrict__ node_state,
    float* __restrict__ path_state, float* __restrict__ m_seg,
    const unsigned short* __restrict__ WxP, const unsigned short* __restrict__ WhP,
    const float* __restrict__ gru_b,
    const int* __restrict__ links_to_paths, const int* __restrict__ nodes_to_paths) {
  __shared__ float sXf[2][16][264];        // x_t f32, double-buffered (async dest)
  __shared__ unsigned short sXb[16][264];  // x_t bf16 (converted once per step)
  __shared__ unsigned short sH[16][136];   // h bf16
  __shared__ float sHf[16][132];           // h f32
  __shared__ float sPX[16][392];           // mx preact (384)
  __shared__ float sPH[16][392];           // mh preact (384)
  __shared__ int sLink[2][16];

  const int tid = threadIdx.x;
  const int lane = tid & 31, wave = tid >> 5;
  const int hlf = lane >> 4, lm = lane & 15;
  const int p0 = blockIdx.x * 16;
  const float* gb0 = gru_b;
  const float* gb1 = gru_b + 384;

  // preload this wave's weight slice into registers (loop-invariant over t)
  v16bf wx[2][8], wh[2][4];
#pragma unroll
  for (int j = 0; j < 2; ++j) {
    int nt = wave * 2 + j;
#pragma unroll
    for (int kt = 0; kt < 8; ++kt) wx[j][kt] = load_frag_glb(WxP, (long)kt * 24 + nt, lane);
#pragma unroll
    for (int kt = 0; kt < 4; ++kt) wh[j][kt] = load_frag_glb(WhP, (long)kt * 24 + nt, lane);
  }

  if (tid < 128) {  // load initial h = path_state
    int p = tid >> 3, c0 = (tid & 7) * 16;
    const float* src = path_state + (size_t)(p0 + p) * 128 + c0;
#pragma unroll
    for (int i = 0; i < 16; i += 2) {
      float v0 = src[i], v1 = src[i + 1];
      sHf[p][c0 + i] = v0; sHf[p][c0 + i + 1] = v1;
      *reinterpret_cast<unsigned*>(&sH[p][c0 + i]) = pk2bf(v0, v1);
    }
  }

  // prologue: async gather of x_0 into buffer 0
  if (tid < 128) {
    int p = tid >> 3, sub = tid & 7;
    int e = (p0 + p) * L_C + 0;
    int li = links_to_paths[e];
    int ni = nodes_to_paths[e];
    if (sub == 0) sLink[0][p] = li;
    const float* ls = link_state + (size_t)li * 128;
    const float* ns = node_state + (size_t)ni * 128;
    const float* dl = &sXf[0][p][0];
#pragma unroll
    for (int q = 0; q < 4; ++q) {
      int cc = sub + 8 * q;                      // 16B chunk index within 512B row
      async_copy16(ls + cc * 4, dl + cc * 4);
      async_copy16(ns + cc * 4, dl + 128 + cc * 4);
    }
  }

#pragma unroll 1
  for (int t = 0; t < L_C; ++t) {
    wait_async0();          // my async writes to buf[t&1] landed
    __syncthreads();        // everyone's writes visible

    // issue async gather for t+1 into the other buffer (overlaps with compute)
    if (t + 1 < L_C && tid < 128) {
      int p = tid >> 3, sub = tid & 7;
      int e = (p0 + p) * L_C + (t + 1);
      int li = links_to_paths[e];
      int ni = nodes_to_paths[e];
      if (sub == 0) sLink[(t + 1) & 1][p] = li;
      const float* ls = link_state + (size_t)li * 128;
      const float* ns = node_state + (size_t)ni * 128;
      const float* dl = &sXf[(t + 1) & 1][p][0];
#pragma unroll
      for (int q = 0; q < 4; ++q) {
        int cc = sub + 8 * q;
        async_copy16(ls + cc * 4, dl + cc * 4);
        async_copy16(ns + cc * 4, dl + 128 + cc * 4);
      }
    }

    // convert x_t f32 -> bf16 exactly once (pk-convert), all fragments then
    // come from cheap b32 LDS loads with zero VALU on the WMMA path
    if (tid < 256) {
      int p = tid >> 4, c0 = (tid & 15) * 16;
      const float* srow = &sXf[t & 1][p][0];
#pragma unroll
      for (int i = 0; i < 16; i += 2)
        *reinterpret_cast<unsigned*>(&sXb[p][c0 + i]) = pk2bf(srow[c0 + i], srow[c0 + i + 1]);
    }
    __syncthreads();

    {  // mx = x @ Wx  (K=256)
      v8f a0 = vzero8(), a1 = vzero8();
#pragma unroll
      for (int kt = 0; kt < 8; ++kt) {
        v16bf a = load_a_lds(&sXb[lm][0], kt * 32 + 8 * hlf);
        a0 = wmma_bf16(a, wx[0][kt], a0);
        a1 = wmma_bf16(a, wx[1][kt], a1);
      }
      int c0 = (wave * 2) * 16 + lm, c1 = (wave * 2 + 1) * 16 + lm;
#pragma unroll
      for (int r = 0; r < 8; ++r) { sPX[r + 8 * hlf][c0] = a0[r]; sPX[r + 8 * hlf][c1] = a1[r]; }
    }
    {  // mh = h @ Wh  (K=128) -- kept separate for reset_after GRU
      v8f a0 = vzero8(), a1 = vzero8();
#pragma unroll
      for (int kt = 0; kt < 4; ++kt) {
        v16bf a = load_a_lds(&sH[lm][0], kt * 32 + 8 * hlf);
        a0 = wmma_bf16(a, wh[0][kt], a0);
        a1 = wmma_bf16(a, wh[1][kt], a1);
      }
      int c0 = (wave * 2) * 16 + lm, c1 = (wave * 2 + 1) * 16 + lm;
#pragma unroll
      for (int r = 0; r < 8; ++r) { sPH[r + 8 * hlf][c0] = a0[r]; sPH[r + 8 * hlf][c1] = a1[r]; }
    }
    __syncthreads();

    if (tid < 256) {  // gates + h update + scatter-sum into m_seg (L2-resident, 5 MB)
      int p = tid >> 4, c0 = (tid & 15) * 8;
      int li = sLink[t & 1][p];
      float hnv[8];
#pragma unroll
      for (int i = 0; i < 8; ++i) {
        int c = c0 + i;
        float xz = sPX[p][c]       + gb0[c];
        float hz = sPH[p][c]       + gb1[c];
        float xr = sPX[p][128 + c] + gb0[128 + c];
        float hr = sPH[p][128 + c] + gb1[128 + c];
        float xh = sPX[p][256 + c] + gb0[256 + c];
        float hh = sPH[p][256 + c] + gb1[256 + c];
        float z = 1.f / (1.f + __expf(-(xz + hz)));
        float r = 1.f / (1.f + __expf(-(xr + hr)));
        float cand = tanhf(xh + r * hh);
        float h = sHf[p][c];
        float hn = z * h + (1.f - z) * cand;
        hnv[i] = hn;
        sHf[p][c] = hn;
        atomicAdd(&m_seg[(size_t)li * 128 + c], hn);
      }
#pragma unroll
      for (int i = 0; i < 8; i += 2)
        *reinterpret_cast<unsigned*>(&sH[p][c0 + i]) = pk2bf(hnv[i], hnv[i + 1]);
    }
    __syncthreads();
  }

  if (tid < 128) {  // write back final h
    int p = tid >> 3, c0 = (tid & 7) * 16;
    float* dst = path_state + (size_t)(p0 + p) * 128 + c0;
#pragma unroll
    for (int i = 0; i < 16; ++i) dst[i] = sHf[p][c0 + i];
  }
}

// ---------------- fused 5-layer edge MLP ----------------

__global__ __launch_bounds__(128) void k_edge_mlp(
    const float* __restrict__ node_state, float* __restrict__ link_state,
    const float* __restrict__ m_seg,
    const unsigned short* __restrict__ W0, const unsigned short* __restrict__ W1,
    const unsigned short* __restrict__ W2, const unsigned short* __restrict__ W3,
    const unsigned short* __restrict__ W4,
    const float* __restrict__ b0, const float* __restrict__ b1, const float* __restrict__ b2,
    const float* __restrict__ b3, const float* __restrict__ b4,
    const int* __restrict__ links_to_nodes) {
  __shared__ unsigned short sX[16][392];
  __shared__ unsigned short sA[16][136];
  __shared__ unsigned short sB[16][136];

  const int tid = threadIdx.x;
  const int lane = tid & 31, wave = tid >> 5;
  const int hlf = lane >> 4, lm = lane & 15;
  const int l0 = blockIdx.x * 16;

  {  // gather concat(node_state[ltn[l]], link_state[l], m_seg[l])
    int p = tid >> 3, seg = tid & 7;
    int l = l0 + p;
    int nd = links_to_nodes[l];
    const float* pa = node_state + (size_t)nd * 128;
    const float* pb = link_state + (size_t)l * 128;
    const float* pc = m_seg + (size_t)l * 128;
    int c0 = seg * 48;
#pragma unroll
    for (int i = 0; i < 48; i += 2) {
      int c = c0 + i;
      float v0 = (c < 128) ? pa[c] : (c < 256) ? pb[c - 128] : pc[c - 256];
      int c2 = c + 1;
      float v1 = (c2 < 128) ? pa[c2] : (c2 < 256) ? pb[c2 - 128] : pc[c2 - 256];
      *reinterpret_cast<unsigned*>(&sX[p][c]) = pk2bf(v0, v1);
    }
  }
  __syncthreads();

  const unsigned short* Wl[5] = {W0, W1, W2, W3, W4};
  const float* Bl[5] = {b0, b1, b2, b3, b4};
  const unsigned short* cur = &sX[0][0];
  int curStride = 392;

  for (int layer = 0; layer < 5; ++layer) {
    int kT = (layer == 0) ? 12 : 4;
    const unsigned short* row = cur + lm * curStride;
    v8f a0 = gemm_acc(row, kT, Wl[layer], 8, wave * 2,     lane, hlf, vzero8());
    v8f a1 = gemm_acc(row, kT, Wl[layer], 8, wave * 2 + 1, lane, hlf, vzero8());
    if (layer < 4) {
      unsigned short* outb = (layer & 1) ? &sB[0][0] : &sA[0][0];
      const float* bias = Bl[layer];
#pragma unroll
      for (int jj = 0; jj < 2; ++jj) {
        v8f acc = jj ? a1 : a0;
        int col = (wave * 2 + jj) * 16 + lm;
        float bv = bias[col];
#pragma unroll
        for (int r = 0; r < 8; ++r) {
          float v = acc[r] + bv;
          v = v > 0.f ? v : 0.f;
          outb[(r + 8 * hlf) * 136 + col] = f2bf(v);
        }
      }
      __syncthreads();
      cur = outb;
      curStride = 136;
    } else {  // final linear layer, f32 out, in-place (own rows only)
      const float* bias = Bl[4];
#pragma unroll
      for (int jj = 0; jj < 2; ++jj) {
        v8f acc = jj ? a1 : a0;
        int col = (wave * 2 + jj) * 16 + lm;
        float bv = bias[col];
#pragma unroll
        for (int r = 0; r < 8; ++r)
          link_state[(size_t)(l0 + r + 8 * hlf) * 128 + col] = acc[r] + bv;
      }
    }
  }
}

// ---------------- GCN ----------------

__global__ void k_agg_scatter(const float* __restrict__ link_state,
                              const int* __restrict__ nodes_to_links,
                              const int* __restrict__ links_to_nodes,
                              float* __restrict__ agg) {
  int i = blockIdx.x * blockDim.x + threadIdx.x;
  if (i >= N_LINKS_C * 128) return;
  int l = i >> 7, c = i & 127;
  float v = link_state[(size_t)nodes_to_links[l] * 128 + c];
  atomicAdd(&agg[(size_t)links_to_nodes[l] * 128 + c], v);
}

// tmp = [node_state | agg] @ gcn_w, emitted directly in packed-B fragment layout
__global__ __launch_bounds__(128) void k_gcn_xw(
    const float* __restrict__ node_state, const float* __restrict__ agg,
    const unsigned short* __restrict__ GcnWP, unsigned short* __restrict__ BpTmp) {
  __shared__ unsigned short sX[16][264];
  const int tid = threadIdx.x;
  const int lane = tid & 31, wave = tid >> 5;
  const int hlf = lane >> 4, lm = lane & 15;
  const int n0 = blockIdx.x * 16;

  {
    int p = tid >> 3, seg = tid & 7;
    int c0 = seg * 32;
    const float* pa = node_state + (size_t)(n0 + p) * 128;
    const float* pb = agg + (size_t)(n0 + p) * 128;
#pragma unroll
    for (int i = 0; i < 32; i += 2) {
      int c = c0 + i;
      float v0 = (c < 128) ? pa[c] : pb[c - 128];
      float v1 = (c + 1 < 128) ? pa[c + 1] : pb[c + 1 - 128];
      *reinterpret_cast<unsigned*>(&sX[p][c]) = pk2bf(v0, v1);
    }
  }
  __syncthreads();

  const unsigned short* row = &sX[lm][0];
#pragma unroll
  for (int jj = 0; jj < 2; ++jj) {
    int nt = wave * 2 + jj;
    v8f acc = gemm_acc(row, 8, GcnWP, 8, nt, lane, hlf, vzero8());
#pragma unroll
    for (int r = 0; r < 8; ++r) {
      int M = n0 + r + 8 * hlf;                 // node row -> K index of next GEMM
      int kt2 = M >> 5, h2 = (M >> 3) & 1, i2 = (M & 7) | ((M >> 1) & 8);
      BpTmp[(size_t)(((kt2 * 8 + nt) * 32) + h2 * 16 + lm) * 16 + i2] = f2bf(acc[r]);
    }
  }
}

// node_state = laplacian @ tmp + gcn_b   (M=2000, K=2016 padded, N=128)
__global__ __launch_bounds__(128) void k_gcn_lap(
    const unsigned short* __restrict__ LapAP, const unsigned short* __restrict__ BpTmp,
    const float* __restrict__ gcn_b, float* __restrict__ node_state) {
  const int tid = threadIdx.x;
  const int lane = tid & 31, wave = tid >> 5;
  const int hlf = lane >> 4, lm = lane & 15;
  int gw = blockIdx.x * 4 + wave;
  int mt = gw >> 3, nt = gw & 7;
  v8f acc = vzero8();
#pragma unroll 2
  for (int kt = 0; kt < 63; ++kt) {
    v16bf a = load_frag_glb(LapAP, (long)mt * 63 + kt, lane);
    v16bf b = load_frag_glb(BpTmp, (long)kt * 8 + nt, lane);
    acc = wmma_bf16(a, b, acc);
  }
  int col = nt * 16 + lm;
  float bv = gcn_b[col];
#pragma unroll
  for (int r = 0; r < 8; ++r)
    node_state[(size_t)(mt * 16 + r + 8 * hlf) * 128 + col] = acc[r] + bv;
}

// ---------------- fused readout ----------------

__global__ __launch_bounds__(128) void k_readout(
    const float* __restrict__ path_state,
    const unsigned short* __restrict__ Ro0P, const unsigned short* __restrict__ Ro1P,
    const float* __restrict__ rb0, const float* __restrict__ rb1,
    const float* __restrict__ fin_w, const float* __restrict__ fin_b,
    float* __restrict__ out) {
  __shared__ unsigned short sPs[16][136];
  __shared__ float sPsF[16][132];
  __shared__ unsigned short sR[16][264];
  __shared__ float sR1[16][264];

  const int tid = threadIdx.x;
  const int lane = tid & 31, wave = tid >> 5;
  const int hlf = lane >> 4, lm = lane & 15;
  const int p0 = blockIdx.x * 16;

  {
    int p = tid >> 3, c0 = (tid & 7) * 16;
    const float* src = path_state + (size_t)(p0 + p) * 128 + c0;
#pragma unroll
    for (int i = 0; i < 16; i += 2) {
      float v0 = src[i], v1 = src[i + 1];
      sPsF[p][c0 + i] = v0; sPsF[p][c0 + i + 1] = v1;
      *reinterpret_cast<unsigned*>(&sPs[p][c0 + i]) = pk2bf(v0, v1);
    }
  }
  __syncthreads();

  {  // r0 = relu(ps @ ro_w0 + b0), K=128 N=256
    const unsigned short* row = &sPs[lm][0];
#pragma unroll
    for (int jj = 0; jj < 4; ++jj) {
      int nt = wave * 4 + jj;
      v8f acc = gemm_acc(row, 4, Ro0P, 16, nt, lane, hlf, vzero8());
      int col = nt * 16 + lm;
      float bv = rb0[col];
#pragma unroll
      for (int r = 0; r < 8; ++r) {
        float v = acc[r] + bv; v = v > 0.f ? v : 0.f;
        sR[r + 8 * hlf][col] = f2bf(v);
      }
    }
  }
  __syncthreads();
  {  // r1 = relu(r0 @ ro_w1 + b1), K=256 N=256
    const unsigned short* row = &sR[lm][0];
#pragma unroll
    for (int jj = 0; jj < 4; ++jj) {
      int nt = wave * 4 + jj;
      v8f acc = gemm_acc(row, 8, Ro1P, 16, nt, lane, hlf, vzero8());
      int col = nt * 16 + lm;
      float bv = rb1[col];
#pragma unroll
      for (int r = 0; r < 8; ++r) {
        float v = acc[r] + bv; v = v > 0.f ? v : 0.f;
        sR1[r + 8 * hlf][col] = v;
      }
    }
  }
  __syncthreads();
  if (tid < 16) {  // o = [r1 | ps] @ fin_w + fin_b
    int p = tid;
    float acc = fin_b[0];
    for (int c = 0; c < 256; ++c) acc += sR1[p][c] * fin_w[c];
    for (int c = 0; c < 128; ++c) acc += sPsF[p][c] * fin_w[256 + c];
    out[p0 + p] = acc;
  }
}

// ---------------- host launcher ----------------

extern "C" void kernel_launch(void* const* d_in, const int* in_sizes, int n_in,
                              void* d_out, int out_size, void* d_ws, size_t ws_size,
                              hipStream_t stream) {
  (void)in_sizes; (void)n_in; (void)out_size; (void)ws_size;
  const float* link_init = (const float*)d_in[0];
  const float* node_init = (const float*)d_in[1];
  const float* path_init = (const float*)d_in[2];
  const float* laplacian = (const float*)d_in[3];
  const float* gru_wx    = (const float*)d_in[4];
  const float* gru_wh    = (const float*)d_in[5];
  const float* gru_b     = (const float*)d_in[6];
  const float* edge_w[5];
  const float* edge_b[5];
  for (int i = 0; i < 5; ++i) { edge_w[i] = (const float*)d_in[7 + 2 * i]; edge_b[i] = (const float*)d_in[8 + 2 * i]; }
  const float* gcn_w = (const float*)d_in[17];
  const float* gcn_b = (const float*)d_in[18];
  const float* ro_w0 = (const float*)d_in[19];
  const float* ro_b0 = (const float*)d_in[20];
  const float* ro_w1 = (const float*)d_in[21];
  const float* ro_b1 = (const float*)d_in[22];
  const float* fin_w = (const float*)d_in[23];
  const float* fin_b = (const float*)d_in[24];
  const int* links_to_paths = (const int*)d_in[27];
  const int* nodes_to_paths = (const int*)d_in[28];
  const int* links_to_nodes = (const int*)d_in[29];
  const int* nodes_to_links = (const int*)d_in[30];

  char* ws = (char*)d_ws;
  size_t off = 0;
  auto carve = [&](size_t bytes) -> void* {
    void* p = ws + off;
    off += (bytes + 255) & ~(size_t)255;
    return p;
  };
  float* link_state = (float*)carve((size_t)N_LINKS_C * 128 * 4);
  float* node_state = (float*)carve((size_t)N_NODES_C * 128 * 4);
  float* path_state = (float*)carve((size_t)N_PATHS_C * 128 * 4);
  float* m_seg      = (float*)carve((size_t)N_LINKS_C * 128 * 4);
  float* agg        = (float*)carve((size_t)N_NODES_C * 128 * 4);
  unsigned short* WxP   = (unsigned short*)carve((size_t)8 * 24 * 32 * 16 * 2);
  unsigned short* WhP   = (unsigned short*)carve((size_t)4 * 24 * 32 * 16 * 2);
  unsigned short* Ew0P  = (unsigned short*)carve((size_t)12 * 8 * 32 * 16 * 2);
  unsigned short* EwP[4];
  for (int i = 0; i < 4; ++i) EwP[i] = (unsigned short*)carve((size_t)4 * 8 * 32 * 16 * 2);
  unsigned short* GcnWP = (unsigned short*)carve((size_t)8 * 8 * 32 * 16 * 2);
  unsigned short* Ro0P  = (unsigned short*)carve((size_t)4 * 16 * 32 * 16 * 2);
  unsigned short* Ro1P  = (unsigned short*)carve((size_t)8 * 16 * 32 * 16 * 2);
  unsigned short* LapAP = (unsigned short*)carve((size_t)125 * 63 * 32 * 16 * 2);
  unsigned short* BpTmp = (unsigned short*)carve((size_t)63 * 8 * 32 * 16 * 2);

  auto cdiv = [](int a, int b) { return (a + b - 1) / b; };

  // pack weights into WMMA fragment layouts (bf16)
  k_pack_b<<<cdiv(8 * 24 * 32, 256), 256, 0, stream>>>(gru_wx, WxP, 8, 24, 256, 384);
  k_pack_b<<<cdiv(4 * 24 * 32, 256), 256, 0, stream>>>(gru_wh, WhP, 4, 24, 128, 384);
  k_pack_b<<<cdiv(12 * 8 * 32, 256), 256, 0, stream>>>(edge_w[0], Ew0P, 12, 8, 384, 128);
  for (int i = 0; i < 4; ++i)
    k_pack_b<<<cdiv(4 * 8 * 32, 256), 256, 0, stream>>>(edge_w[i + 1], EwP[i], 4, 8, 128, 128);
  k_pack_b<<<cdiv(8 * 8 * 32, 256), 256, 0, stream>>>(gcn_w, GcnWP, 8, 8, 256, 128);
  k_pack_b<<<cdiv(4 * 16 * 32, 256), 256, 0, stream>>>(ro_w0, Ro0P, 4, 16, 128, 256);
  k_pack_b<<<cdiv(8 * 16 * 32, 256), 256, 0, stream>>>(ro_w1, Ro1P, 8, 16, 256, 256);
  k_pack_a_lap<<<cdiv(125 * 63 * 32, 256), 256, 0, stream>>>(laplacian, LapAP);

  // initial states
  k_init_vec_state<<<cdiv(N_LINKS_C * 128, 256), 256, 0, stream>>>(link_init, link_state, N_LINKS_C);
  k_init_vec_state<<<cdiv(N_NODES_C * 128, 256), 256, 0, stream>>>(node_init, node_state, N_NODES_C);
  k_init_path_state<<<cdiv(N_PATHS_C * 128, 256), 256, 0, stream>>>(path_init, path_state);

  for (int it = 0; it < 4; ++it) {
    k_zero_f32<<<cdiv(N_LINKS_C * 128, 256), 256, 0, stream>>>(m_seg, N_LINKS_C * 128);
    k_gru_scan<<<N_PATHS_C / 16, 384, 0, stream>>>(link_state, node_state, path_state, m_seg,
                                                   WxP, WhP, gru_b, links_to_paths, nodes_to_paths);
    k_edge_mlp<<<N_LINKS_C / 16, 128, 0, stream>>>(node_state, link_state, m_seg,
                                                   Ew0P, EwP[0], EwP[1], EwP[2], EwP[3],
                                                   edge_b[0], edge_b[1], edge_b[2], edge_b[3], edge_b[4],
                                                   links_to_nodes);
    k_zero_f32<<<cdiv(N_NODES_C * 128, 256), 256, 0, stream>>>(agg, N_NODES_C * 128);
    k_agg_scatter<<<cdiv(N_LINKS_C * 128, 256), 256, 0, stream>>>(link_state, nodes_to_links,
                                                                  links_to_nodes, agg);
    k_zero_f32<<<cdiv(63 * 8 * 32 * 16 * 2 / 4, 256), 256, 0, stream>>>((float*)BpTmp,
                                                                        63 * 8 * 32 * 16 * 2 / 4);
    k_gcn_xw<<<N_NODES_C / 16, 128, 0, stream>>>(node_state, agg, GcnWP, BpTmp);
    k_gcn_lap<<<250, 128, 0, stream>>>(LapAP, BpTmp, gcn_b, node_state);
  }
  k_readout<<<N_PATHS_C / 16, 128, 0, stream>>>(path_state, Ro0P, Ro1P, ro_b0, ro_b1,
                                                fin_w, fin_b, (float*)d_out);
}
